// AdaptiveConv_55851754717599
// MI455X (gfx1250) — compile-verified, matrix operand order checked by reading, not run on hardware
//
#include <hip/hip_runtime.h>

// ---------------------------------------------------------------------------
// AdaptiveConv on MI455X (gfx1250):
//   out[b,o,ho,wo] = sum_{kh,kw} sum_i W[o,i,kh,kw] * bilin(x[b,i], yy, xx)
// Inner contraction done with v_wmma_f32_16x16x32_bf16 (f32 accumulate).
// One workgroup = 4 waves = (b, ho, 16-wide pixel tile); wave w owns
// c_out group w (16 rows). Samples staged in LDS as packed bf16 pairs.
// ---------------------------------------------------------------------------

typedef __attribute__((ext_vector_type(16))) __bf16    v16bf;
typedef __attribute__((ext_vector_type(8)))  float     v8f;
typedef __attribute__((ext_vector_type(4)))  unsigned  v4u;
typedef __attribute__((ext_vector_type(8)))  unsigned  v8u;

#define CIN   64
#define COUT  64
#define HH    256
#define WW    256
#define OH    256
#define OW    256
#define PLANE 65536  // H*W

__device__ __forceinline__ unsigned pack_bf16(float a, float b) {
  unsigned ua = __builtin_bit_cast(unsigned, a);
  unsigned ub = __builtin_bit_cast(unsigned, b);
  ua = (ua + 0x7FFFu + ((ua >> 16) & 1u)) >> 16;   // RNE f32 -> bf16
  ub = (ub + 0x7FFFu + ((ub >> 16) & 1u)) >> 16;
  return ua | (ub << 16);
}

// ---------------------------------------------------------------------------
// Repack weight[o,i,kh,kw] (f32) into per-lane bf16 A-fragments.
// Fragment f = (g*9 + tap)*2 + kf  (g = c_out group 0..3, kf = K-half 0..1).
// 16-bit A 16x32 ISA layout: lane<16 -> M=lane, K pairs {0,2,4,6}+{16..22};
// lane>=16 -> M=lane-16, K pairs {8..14}+{24..30}. One u32 = bf16 pair (K,K+1).
// Fragment storage: rp[f*256 + lane*8 + d]  -> wave loads 32B/lane (b128 x2).
// ---------------------------------------------------------------------------
__global__ void repack_weights_kernel(const float* __restrict__ w,
                                      unsigned* __restrict__ rp) {
  const int f    = blockIdx.x;       // 0..71
  const int t    = threadIdx.x;      // 0..255
  const int lane = t >> 3;
  const int d    = t & 7;
  const int kf   = f & 1;
  const int tap  = (f >> 1) % 9;
  const int g    = f / 18;
  const int kh   = tap / 3, kw = tap % 3;
  const int M    = lane & 15;
  const int kOff = (lane < 16) ? 0 : 8;
  const int K    = kf * 32 + kOff + ((d < 4) ? 2 * d : 16 + 2 * (d - 4));
  const int o    = g * 16 + M;
  const float w0 = w[((o * CIN + K    ) * 3 + kh) * 3 + kw];
  const float w1 = w[((o * CIN + K + 1) * 3 + kh) * 3 + kw];
  rp[f * 256 + lane * 8 + d] = pack_bf16(w0, w1);
}

// ---------------------------------------------------------------------------
// Main kernel. grid = B * OH * (OW/16) blocks of 128 threads (4 waves).
// ---------------------------------------------------------------------------
__global__ __launch_bounds__(128)
void adaptive_conv_kernel(const float* __restrict__ xin,
                          const float* __restrict__ sh,
                          const float* __restrict__ sw,
                          const float* __restrict__ dil,
                          const unsigned* __restrict__ rp,
                          float* __restrict__ out) {
  // LDS sample tile: [pixel 0..15][chan-pair 0..31] packed bf16 (2 KB)
  __shared__ unsigned lds[16 * 32];

  const int wg  = blockIdx.x;
  const int b   = wg >> 12;            // / (256 rows * 16 tiles)
  const int ho  = (wg >> 4) & 255;
  const int wo0 = (wg & 15) << 4;

  const int t    = threadIdx.x;
  const int lane = t & 31;
  const int wv   = t >> 5;             // wave id == c_out group
  const int p    = t & 15;             // sampling pixel
  const int cb   = (t >> 4) << 3;      // sampling channel base (0..56)

  // Per-pixel sampling state
  const float sy = (sh[b * OH + ho]       + 1.0f) * 0.5f * (float)(HH - 1);
  const float sx = (sw[b * OW + wo0 + p]  + 1.0f) * 0.5f * (float)(WW - 1);
  const float dl = dil[(b * OH + ho) * OW + wo0 + p];
  const float* xb = xin + ((size_t)b * CIN + cb) * PLANE;

  // B-fragment read addressing (per lane)
  const int  n    = lane & 15;
  const int  hsel = lane >> 4;                     // 0: K lo-half, 1: hi-half
  const unsigned* bbase = &lds[n * 32 + hsel * 4];

  v8f acc = {};

#pragma unroll
  for (int tap = 0; tap < 9; ++tap) {
    const int kh = tap / 3, kw = tap % 3;

    // ---- bilinear setup (once per thread per tap) ----
    const float yy  = sy + (float)(kh - 1) * dl;
    const float xx  = sx + (float)(kw - 1) * dl;
    const float y0f = floorf(yy), x0f = floorf(xx);
    const float wy1 = yy - y0f,   wx1 = xx - x0f;
    const float wy0 = 1.0f - wy1, wx0 = 1.0f - wx1;
    const int y0 = (int)y0f, x0 = (int)x0f;
    const int y1 = y0 + 1,   x1 = x0 + 1;
    const float my0 = (y0 >= 0 && y0 < HH) ? 1.0f : 0.0f;
    const float my1 = (y1 >= 0 && y1 < HH) ? 1.0f : 0.0f;
    const float mx0 = (x0 >= 0 && x0 < WW) ? 1.0f : 0.0f;
    const float mx1 = (x1 >= 0 && x1 < WW) ? 1.0f : 0.0f;
    const int y0c = min(max(y0, 0), HH - 1), y1c = min(max(y1, 0), HH - 1);
    const int x0c = min(max(x0, 0), WW - 1), x1c = min(max(x1, 0), WW - 1);
    const float W00 = wy0 * wx0 * my0 * mx0;
    const float W01 = wy0 * wx1 * my0 * mx1;
    const float W10 = wy1 * wx0 * my1 * mx0;
    const float W11 = wy1 * wx1 * my1 * mx1;
    const int off00 = y0c * WW + x0c, off01 = y0c * WW + x1c;
    const int off10 = y1c * WW + x0c, off11 = y1c * WW + x1c;

    // ---- gather 8 channels for this pixel ----
    float v[8];
#pragma unroll
    for (int c = 0; c < 8; ++c) {
      const float* pc = xb + (size_t)c * PLANE;
      v[c] = W00 * pc[off00] + W01 * pc[off01] +
             W10 * pc[off10] + W11 * pc[off11];
    }
    v4u pk;
    pk.x = pack_bf16(v[0], v[1]);
    pk.y = pack_bf16(v[2], v[3]);
    pk.z = pack_bf16(v[4], v[5]);
    pk.w = pack_bf16(v[6], v[7]);

    __syncthreads();                              // prior tap's B reads done
    *(v4u*)&lds[p * 32 + (cb >> 1)] = pk;         // ds_write_b128
    __syncthreads();                              // sample tile visible

    // ---- A fragments for this wave's c_out group (L2-hot, b128 loads) ----
    const unsigned* ap = rp + (size_t)((wv * 9 + tap) * 2) * 256 + lane * 8;
    const v8u a0u = *(const v8u*)ap;              // K = 0..31
    const v8u a1u = *(const v8u*)(ap + 256);      // K = 32..63
    const v16bf A0 = __builtin_bit_cast(v16bf, a0u);
    const v16bf A1 = __builtin_bit_cast(v16bf, a1u);

    // ---- B fragments from LDS (ds_read_b128 x4) ----
    const v4u b0lo = *(const v4u*)(bbase);        // kf=0, K pairs 0..3/4..7
    const v4u b0hi = *(const v4u*)(bbase + 8);    // kf=0, pairs 8..11/12..15
    const v4u b1lo = *(const v4u*)(bbase + 16);   // kf=1
    const v4u b1hi = *(const v4u*)(bbase + 24);
    const v8u b0u = __builtin_shufflevector(b0lo, b0hi, 0, 1, 2, 3, 4, 5, 6, 7);
    const v8u b1u = __builtin_shufflevector(b1lo, b1hi, 0, 1, 2, 3, 4, 5, 6, 7);
    const v16bf B0 = __builtin_bit_cast(v16bf, b0u);
    const v16bf B1 = __builtin_bit_cast(v16bf, b1u);

    // ---- D = A*B + C, f32 accumulate ----
    acc = __builtin_amdgcn_wmma_f32_16x16x32_bf16(
        false, A0, false, B0, (short)0, acc, false, false);
    acc = __builtin_amdgcn_wmma_f32_16x16x32_bf16(
        false, A1, false, B1, (short)0, acc, false, false);
  }

  // ---- store: C/D layout VGPR r -> M = r (+8 for lanes 16..31), lane -> N ----
  const int row = wv * 16 + hsel * 8;
  float* ob = out + (((size_t)b * COUT + row) * OH + ho) * OW + wo0 + n;
#pragma unroll
  for (int r = 0; r < 8; ++r)
    ob[(size_t)r * (OH * OW)] = acc[r];
}

extern "C" void kernel_launch(void* const* d_in, const int* in_sizes, int n_in,
                              void* d_out, int out_size, void* d_ws, size_t ws_size,
                              hipStream_t stream) {
  const float* x  = (const float*)d_in[0];  // [8,64,256,256]
  const float* sh = (const float*)d_in[1];  // [8,256]
  const float* sw = (const float*)d_in[2];  // [8,256]
  const float* dl = (const float*)d_in[3];  // [8,1,256,256]
  const float* w  = (const float*)d_in[4];  // [64,64,3,3]

  unsigned* rp = (unsigned*)d_ws;           // 72 frags * 256 u32 = 73728 B

  repack_weights_kernel<<<72, 256, 0, stream>>>(w, rp);
  adaptive_conv_kernel<<<8 * 256 * 16, 128, 0, stream>>>(
      x, sh, sw, dl, rp, (float*)d_out);
}
